// ActorLayerGAT_81604378624407
// MI455X (gfx1250) — matrix operand order repaired and checked by reading, not compiled, not run on hardware
//
#include <hip/hip_runtime.h>

typedef __bf16 bf16;
typedef __attribute__((ext_vector_type(16))) __bf16 v16bf;
typedef __attribute__((ext_vector_type(8)))  __bf16 v8bf;
typedef __attribute__((ext_vector_type(8)))  float  v8f;

#define D     512
#define HEADS 4
#define CH    128
#define NN    5
#define NE    10
#define TB    16
#define NK    16   // K steps of 32 (D/32)

// ---------------------------------------------------------------------------
// WMMA helpers (CDNA5 wave32, V_WMMA_F32_16X16X32_BF16)
// ---------------------------------------------------------------------------
__device__ __forceinline__ v8f wmma_bf16(v16bf a, v16bf b, v8f c) {
  return __builtin_amdgcn_wmma_f32_16x16x32_bf16(false, a, false, b, (short)0, c,
                                                 false, false);
}

// A fragment: 16x32 bf16, lanes 0-15 hold row M=lane, K = kt*32 + {0..15};
// lanes 16-31 hold row M=lane-16, K = kt*32 + {16..31}.
__device__ __forceinline__ v16bf load_A_frag(const bf16* Abuf, int kt, int lane) {
  int row  = lane & 15;
  int koff = kt * 32 + ((lane >> 4) << 4);
  const bf16* p = Abuf + row * D + koff;
  union { v16bf v; v8bf h[2]; } u;
  u.h[0] = *(const v8bf*)(p);
  u.h[1] = *(const v8bf*)(p + 8);
  return u.v;
}

// B fragment: pre-swizzled by repack kernel; lane's 16 bf16 are contiguous.
__device__ __forceinline__ v16bf load_B_frag(const bf16* Wq, int nt, int kt, int lane) {
  const bf16* p = Wq + (((size_t)(nt * NK + kt) * 32 + lane) << 4);
  union { v16bf v; v8bf h[2]; } u;
  u.h[0] = *(const v8bf*)(p);
  u.h[1] = *(const v8bf*)(p + 8);
  return u.v;
}

// ---------------------------------------------------------------------------
// Pre-kernel: convert f32 weights -> bf16 in WMMA B-fragment order.
// B-fragment layout (32x16 bf16): lane = (kk>=16)*16 + (n&15), elem e = kk&15.
// ---------------------------------------------------------------------------
__global__ void repack_weights(const float* __restrict__ Wc,
                               const float* __restrict__ Wl,
                               const float* __restrict__ Wr,
                               bf16* __restrict__ out) {
  int idx = blockIdx.x * blockDim.x + threadIdx.x;   // 3 * 512 * 512 total
  int w = idx >> 18;
  int r = idx & 262143;
  int k = r >> 9, n = r & 511;
  const float* W = (w == 0) ? Wc : ((w == 1) ? Wl : Wr);
  float v = W[r];
  int ktile = k >> 5, kk = k & 31;
  int ntile = n >> 4;
  int lane  = ((kk >> 4) << 4) | (n & 15);
  int e     = kk & 15;
  out[(size_t)w * 262144 + ((((size_t)(ntile * NK + ktile)) * 32 + lane) << 4) + e] =
      (bf16)v;
}

// ---------------------------------------------------------------------------
// Fused GATv2 actor layer: 1 block = 16 batch rows, 8 wave32 waves.
// All 5 node activations staged at once -> each Wl/Wr B-fragment feeds 5 WMMAs.
// ---------------------------------------------------------------------------
__global__ __launch_bounds__(256, 1)
void gat_fused(const float* __restrict__ in0, const float* __restrict__ in1,
               const float* __restrict__ in2, const float* __restrict__ in3,
               const float* __restrict__ in4, const float* __restrict__ h_state,
               const float* __restrict__ bc,  const float* __restrict__ bl,
               const float* __restrict__ br,  const float* __restrict__ att,
               const float* __restrict__ bias_out,
               const float* __restrict__ gamma, const float* __restrict__ beta,
               const int*   __restrict__ eidx,
               const bf16*  __restrict__ Wcq, const bf16* __restrict__ Wlq,
               const bf16*  __restrict__ Wrq,
               float* __restrict__ out, int Btot) {
  __shared__ __align__(16) bf16  Abuf5[NN * TB * D];    // 80 KB (node activations)
  __shared__ __align__(16) float ctx[TB * D];           // 32 KB (reused as pooled)
  __shared__ __align__(16) bf16  XL[TB * NN * D];       // 80 KB
  __shared__ __align__(16) bf16  XR[TB * NN * D];       // 80 KB
  __shared__ float logits[TB * NE * HEADS];             // 2.5 KB
  __shared__ float alpha_s[TB * NE * HEADS];            // 2.5 KB
  __shared__ float att_s[HEADS * CH];                   // 2 KB
  __shared__ float mden[TB * NN * HEADS * 2];           // 2.5 KB
  __shared__ float red[TB * 16 * 2];                    // 2 KB
  __shared__ int   es[2 * NE];

  const int tid  = threadIdx.x;
  const int wave = tid >> 5;
  const int lane = tid & 31;
  const int row0 = blockIdx.x * TB;

  // ---- stage 0: stage h_state tile as bf16 into Abuf5[0]; cache att, edges ----
  for (int i = tid; i < TB * D; i += 256) {
    int r = i >> 9, c = i & (D - 1);
    Abuf5[i] = (bf16)h_state[(size_t)(row0 + r) * D + c];
  }
  for (int i = tid; i < HEADS * CH; i += 256) att_s[i] = att[i];
  if (tid < 2 * NE) es[tid] = eidx[tid];
  __syncthreads();

  // ---- stage 1: ctx = h_state @ Wc + bc  (each wave: 4 column tiles) ----
  {
    v8f acc[4] = {};
    for (int kt = 0; kt < NK; ++kt) {
      v16bf a = load_A_frag(Abuf5, kt, lane);
#pragma unroll
      for (int j = 0; j < 4; ++j)
        acc[j] = wmma_bf16(a, load_B_frag(Wcq, wave * 4 + j, kt, lane), acc[j]);
    }
#pragma unroll
    for (int j = 0; j < 4; ++j) {
      int col = (wave * 4 + j) * 16 + (lane & 15);
      int rb  = (lane >> 4) << 3;
      float bias = bc[col];
#pragma unroll
      for (int r = 0; r < 8; ++r) ctx[((rb + r) << 9) + col] = acc[j][r] + bias;
    }
  }
  __syncthreads();

  // ---- stage 2a: stage ALL node activations: Abuf5[n] = bf16(input_n + ctx) ----
  {
    const float* ins[NN] = {in0, in1, in2, in3, in4};
#pragma unroll
    for (int node = 0; node < NN; ++node) {
      const float* inp = ins[node];
      for (int i = tid; i < TB * D; i += 256) {
        int r = i >> 9, c = i & (D - 1);
        Abuf5[node * (TB * D) + i] =
            (bf16)(inp[(size_t)(row0 + r) * D + c] + ctx[i]);
      }
    }
  }
  __syncthreads();

  // ---- stage 2b: fused 5-node GEMM: each Wl/Wr fragment drives 5 WMMAs ----
  {
    v8f accL[NN][4] = {}, accR[NN][4] = {};
    for (int kt = 0; kt < NK; ++kt) {
      v16bf a[NN];
#pragma unroll
      for (int n = 0; n < NN; ++n)
        a[n] = load_A_frag(Abuf5 + n * (TB * D), kt, lane);
#pragma unroll
      for (int j = 0; j < 4; ++j) {
        int nt = wave * 4 + j;
        v16bf bL = load_B_frag(Wlq, nt, kt, lane);
        v16bf bR = load_B_frag(Wrq, nt, kt, lane);
#pragma unroll
        for (int n = 0; n < NN; ++n) {
          accL[n][j] = wmma_bf16(a[n], bL, accL[n][j]);
          accR[n][j] = wmma_bf16(a[n], bR, accR[n][j]);
        }
      }
    }
    // write XL, XR with biases
#pragma unroll
    for (int j = 0; j < 4; ++j) {
      int col = (wave * 4 + j) * 16 + (lane & 15);
      int rb  = (lane >> 4) << 3;
      float bL = bl[col], bR = br[col];
#pragma unroll
      for (int n = 0; n < NN; ++n) {
#pragma unroll
        for (int r = 0; r < 8; ++r) {
          int row = rb + r;
          XL[((row * NN + n) << 9) + col] = (bf16)(accL[n][j][r] + bL);
          XR[((row * NN + n) << 9) + col] = (bf16)(accR[n][j][r] + bR);
        }
      }
    }
  }
  __syncthreads();

  // ---- stage 3a: logits[row][e][h] = LeakyReLU(xl[src]+xr[dst]) . att[h] ----
  for (int t = tid; t < TB * NE * HEADS; t += 256) {
    int row = t / (NE * HEADS);
    int rem = t - row * (NE * HEADS);
    int e = rem >> 2, h = rem & 3;
    const bf16* pl = &XL[((row * NN + es[e]) << 9) + h * CH];
    const bf16* pr = &XR[((row * NN + es[NE + e]) << 9) + h * CH];
    const float* pa = &att_s[h * CH];
    float s = 0.f;
    for (int c = 0; c < CH; ++c) {
      float v = (float)pl[c] + (float)pr[c];
      v = (v > 0.f) ? v : 0.2f * v;
      s += v * pa[c];
    }
    logits[t] = s;
  }
  __syncthreads();

  // ---- stage 3b: per-(row, dst node, head) max + denom ----
  for (int t = tid; t < TB * NN * HEADS; t += 256) {
    int row = t / (NN * HEADS);
    int rem = t - row * (NN * HEADS);
    int n = rem >> 2, h = rem & 3;
    float m = -3.0e38f;
#pragma unroll
    for (int e = 0; e < NE; ++e)
      if (es[NE + e] == n) m = fmaxf(m, logits[row * 40 + e * 4 + h]);
    float den = 0.f;
#pragma unroll
    for (int e = 0; e < NE; ++e)
      if (es[NE + e] == n) den += __expf(logits[row * 40 + e * 4 + h] - m);
    mden[t * 2]     = m;
    mden[t * 2 + 1] = den;
  }
  __syncthreads();

  // ---- stage 3c: alpha ----
  for (int t = tid; t < TB * NE * HEADS; t += 256) {
    int row = t / 40;
    int rem = t - row * 40;
    int e = rem >> 2, h = rem & 3;
    int mi = (row * NN + es[NE + e]) * HEADS + h;
    alpha_s[t] = __expf(logits[t] - mden[mi * 2]) / mden[mi * 2 + 1];
  }
  __syncthreads();

  // ---- stage 3d: pooled[row][d] = (1/5) * sum_e alpha * xl[src] + bias_out ----
  for (int i = tid; i < TB * D; i += 256) {
    int row = i >> 9, d = i & (D - 1);
    int h = d >> 7;
    float s = 0.f;
#pragma unroll
    for (int e = 0; e < NE; ++e)
      s += alpha_s[row * 40 + e * 4 + h] * (float)XL[((row * NN + es[e]) << 9) + d];
    ctx[i] = 0.2f * s + bias_out[d];   // ctx reused as pooled
  }
  __syncthreads();

  // ---- stage 3e: LayerNorm over D per row ----
  {
    int row = tid >> 4, sub = tid & 15;
    float s = 0.f, s2 = 0.f;
    const float* p = &ctx[(row << 9) + sub * 32];
    for (int j = 0; j < 32; ++j) { float v = p[j]; s += v; s2 += v * v; }
    red[(row * 16 + sub) * 2]     = s;
    red[(row * 16 + sub) * 2 + 1] = s2;
  }
  __syncthreads();
  if (tid < TB) {
    float s = 0.f, s2 = 0.f;
    for (int j = 0; j < 16; ++j) {
      s  += red[(tid * 16 + j) * 2];
      s2 += red[(tid * 16 + j) * 2 + 1];
    }
    float mu  = s * (1.f / 512.f);
    float var = s2 * (1.f / 512.f) - mu * mu;
    mden[tid * 2]     = mu;                    // mden free now; holds mu/rstd
    mden[tid * 2 + 1] = rsqrtf(var + 1e-5f);
  }
  __syncthreads();
  for (int i = tid; i < TB * D; i += 256) {
    int row = i >> 9, d = i & (D - 1);
    float v = (ctx[i] - mden[row * 2]) * mden[row * 2 + 1];
    out[(size_t)(row0 + row) * D + d] = gamma[d] * v + beta[d];
  }

  // ---- second output: alpha of batch row 0, shape [E, H] ----
  if (blockIdx.x == 0 && tid < NE * HEADS)
    out[(size_t)Btot * D + tid] = alpha_s[tid];   // row 0 => indices e*4+h
}

// ---------------------------------------------------------------------------
extern "C" void kernel_launch(void* const* d_in, const int* in_sizes, int n_in,
                              void* d_out, int out_size, void* d_ws, size_t ws_size,
                              hipStream_t stream) {
  const float* in0      = (const float*)d_in[0];   // striker_identity
  const float* in1      = (const float*)d_in[1];   // striker_state
  const float* in2      = (const float*)d_in[2];   // bowler_identity
  const float* in3      = (const float*)d_in[3];   // bowler_state
  const float* in4      = (const float*)d_in[4];   // partnership
  const float* h_state  = (const float*)d_in[5];
  const float* Wc       = (const float*)d_in[6];
  const float* bc       = (const float*)d_in[7];
  const float* Wl       = (const float*)d_in[8];
  const float* bl       = (const float*)d_in[9];
  const float* Wr       = (const float*)d_in[10];
  const float* br       = (const float*)d_in[11];
  const float* att      = (const float*)d_in[12];
  const float* bias_out = (const float*)d_in[13];
  const float* gamma    = (const float*)d_in[14];
  const float* beta     = (const float*)d_in[15];
  const int*   eidx     = (const int*)d_in[16];

  bf16* Wq = (bf16*)d_ws;                       // 3 * 512KB bf16 fragments
  const int B = in_sizes[0] / D;                // 32768

  repack_weights<<<(3 * 512 * 512) / 256, 256, 0, stream>>>(Wc, Wl, Wr, Wq);

  gat_fused<<<B / TB, 256, 0, stream>>>(
      in0, in1, in2, in3, in4, h_state, bc, bl, br, att, bias_out, gamma, beta,
      eidx, Wq, Wq + 262144, Wq + 524288, (float*)d_out, B);
}
